// IndRNNModel_77549929497105
// MI455X (gfx1250) — matrix operand order repaired and checked by reading, not compiled, not run on hardware
//
#include <hip/hip_runtime.h>
#include <cstdint>

// ---------------------------------------------------------------------------
// IndRNN on gfx1250: proj = x @ W_ih^T fused into the T-scan with bf16 WMMA.
// B=256, T=2048, D=128, H=512, O=128.
//
// Structure (latency-oriented — the scan is 2048 *serial* steps):
//   pass 0: x fp32 -> bf16 into d_ws (one bandwidth-bound sweep)
//   pass 1: scan. One wave per 16x16 (batch x hidden) output tile, NO LDS,
//           NO barriers. W_ih fragment / u / b_ih / h-state register-resident.
//           A-fragments ping-pong in registers, loaded one step ahead so the
//           per-step critical path is just the 4-deep WMMA chain + h update.
//   pass 2: logits = h_last @ W_fc^T + b_fc, row softmax (WMMA over K=512).
// ---------------------------------------------------------------------------

typedef __bf16       v16bf __attribute__((ext_vector_type(16)));
typedef float        v8f   __attribute__((ext_vector_type(8)));
typedef unsigned int v4u   __attribute__((ext_vector_type(4)));

static constexpr int Bn = 256, Tn = 2048, Dn = 128, Hn = 512, On = 128;

// round-to-nearest-even fp32 -> bf16, packed pair (lo = first K)
__device__ __forceinline__ unsigned int pack_bf16x2(float lo, float hi) {
  unsigned int xl = __float_as_uint(lo);
  unsigned int xh = __float_as_uint(hi);
  xl = xl + 0x7FFFu + ((xl >> 16) & 1u);
  xh = xh + 0x7FFFu + ((xh >> 16) & 1u);
  return (xl >> 16) | (xh & 0xFFFF0000u);
}

union Frag16 {
  v16bf v;
  unsigned int u[8];
  v4u q[2];
};

// ---------------------------------------------------------------------------
// Pass 0: x (fp32, 268 MB) -> bf16 (134 MB) in workspace. 8 floats / thread.
// ---------------------------------------------------------------------------
__global__ __launch_bounds__(256) void convert_bf16_kernel(
    const float* __restrict__ x, unsigned short* __restrict__ xb) {
  size_t i8 = (size_t)blockIdx.x * 256 + threadIdx.x;  // index in 8-float units
  const float4* p = (const float4*)x + i8 * 2;
  float4 f0 = p[0], f1 = p[1];
  v4u o = {pack_bf16x2(f0.x, f0.y), pack_bf16x2(f0.z, f0.w),
           pack_bf16x2(f1.x, f1.y), pack_bf16x2(f1.z, f1.w)};
  ((v4u*)xb)[i8] = o;
}

// ---------------------------------------------------------------------------
// Pass 1: fused projection + IndRNN scan.
//   grid = (B/16, H/16) = (16, 32), block = 32 threads (one wave, one tile).
//   PRE = true : read pre-converted bf16 x (no ALU conversion in the loop)
//   PRE = false: read fp32 x, convert in-loop (workspace-too-small fallback)
// ---------------------------------------------------------------------------
template <bool PRE>
__global__ __launch_bounds__(32) void indrnn_scan_kernel(
    const float* __restrict__ x, const unsigned short* __restrict__ xb,
    const float* __restrict__ W_ih, const float* __restrict__ b_ih,
    const float* __restrict__ u_diag, float* __restrict__ h_out) {
  const int lane   = threadIdx.x & 31;
  const int b_base = blockIdx.x * 16;
  const int h_base = blockIdx.y * 16;
  const int nlo    = lane & 15;  // A row (batch) == B col (hidden) selector
  const int khalf  = lane >> 4;  // K-half selector (ISA 16-bit A/B layouts)

  // ---- B fragments: W_ih rows, fp32 -> bf16 once, K = 0..127 ----
  Frag16 bfrag[4];
  {
    const float* wrow = W_ih + (size_t)(h_base + nlo) * Dn;
    #pragma unroll
    for (int kc = 0; kc < 4; ++kc) {
      const float* p = wrow + kc * 32 + khalf * 8;
      #pragma unroll
      for (int i = 0; i < 8; ++i)
        bfrag[kc].u[i] = pack_bf16x2(p[2 * i + (i >= 4 ? 8 : 0)], p[2 * i + 1 + (i >= 4 ? 8 : 0)]);
    }
  }

  const float uval = u_diag[h_base + nlo];
  const float bval = b_ih[h_base + nlo];

  float h[8];
  #pragma unroll
  for (int r = 0; r < 8; ++r) h[r] = 0.f;

  // per-lane x row base (row m = nlo of this b-tile)
  const size_t rowoff = (size_t)(b_base + nlo) * Tn * Dn;
  const float* xl = x + rowoff;
  const v4u* xbl = (const v4u*)(xb + rowoff);  // 16 x uint4 per timestep row

  // load the A fragment (16x32 bf16 per chunk) for timestep t
  auto loadA = [&](int t, Frag16(&A)[4]) {
    if constexpr (PRE) {
      const v4u* p = xbl + (size_t)t * (Dn / 8);
      #pragma unroll
      for (int kc = 0; kc < 4; ++kc) {
        A[kc].q[0] = p[kc * 4 + khalf];      // K = kc*32 + khalf*8 + 0..7
        A[kc].q[1] = p[kc * 4 + 2 + khalf];  //        ... + 16 .. 23
      }
    } else {
      const float4* p = (const float4*)(xl + (size_t)t * Dn);
      #pragma unroll
      for (int kc = 0; kc < 4; ++kc) {
        float4 f0 = p[kc * 8 + khalf * 2];
        float4 f1 = p[kc * 8 + khalf * 2 + 1];
        float4 f2 = p[kc * 8 + 4 + khalf * 2];
        float4 f3 = p[kc * 8 + 4 + khalf * 2 + 1];
        A[kc].u[0] = pack_bf16x2(f0.x, f0.y); A[kc].u[1] = pack_bf16x2(f0.z, f0.w);
        A[kc].u[2] = pack_bf16x2(f1.x, f1.y); A[kc].u[3] = pack_bf16x2(f1.z, f1.w);
        A[kc].u[4] = pack_bf16x2(f2.x, f2.y); A[kc].u[5] = pack_bf16x2(f2.z, f2.w);
        A[kc].u[6] = pack_bf16x2(f3.x, f3.y); A[kc].u[7] = pack_bf16x2(f3.z, f3.w);
      }
    }
  };

  // one IndRNN step: acc = A @ B (K=128), h = relu(acc + b + u*h)
  auto step = [&](const Frag16(&A)[4]) {
    v8f acc = {};
    #pragma unroll
    for (int kc = 0; kc < 4; ++kc)
      acc = __builtin_amdgcn_wmma_f32_16x16x32_bf16(
          false, A[kc].v, false, bfrag[kc].v, (short)0, acc, false, false);
    #pragma unroll
    for (int r = 0; r < 8; ++r) {
      float hn = acc[r] + bval + uval * h[r];
      h[r] = hn > 0.f ? hn : 0.f;
    }
  };

  Frag16 A0[4], A1[4];
  loadA(0, A0);
  for (int t = 0; t < Tn; t += 2) {
    // stay ~1 iteration ahead on loads; warm WGP$ ~8 steps ahead
    loadA(t + 1, A1);
    {
      int tp = (t + 9 < Tn) ? t + 9 : Tn - 1;
      if constexpr (PRE)
        __builtin_prefetch((const void*)(xb + rowoff + (size_t)tp * Dn), 0, 0);
      else
        __builtin_prefetch((const void*)(xl + (size_t)tp * Dn), 0, 0);
    }
    step(A0);
    int t2 = (t + 2 < Tn) ? t + 2 : Tn - 1;
    loadA(t2, A0);
    step(A1);
  }

  // h_last tile: element (m, n) -> [b_base+m, h_base+n]
  #pragma unroll
  for (int r = 0; r < 8; ++r) {
    int m = khalf * 8 + r;
    h_out[(size_t)(b_base + m) * Hn + h_base + nlo] = h[r];
  }
}

// ---------------------------------------------------------------------------
// Pass 2: logits = h_last @ W_fc^T + b_fc, then row softmax.
//   grid = B/16 = 16 blocks, block = 256 threads (8 waves; wave = o-tile).
// ---------------------------------------------------------------------------
__global__ __launch_bounds__(256) void fc_softmax_kernel(
    const float* __restrict__ h_in, const float* __restrict__ W_fc,
    const float* __restrict__ b_fc, float* __restrict__ out) {
  __shared__ float logits[16 * On];

  const int tid    = threadIdx.x;
  const int lane   = tid & 31;
  const int wave   = tid >> 5;  // o-tile index, 8 tiles cover O=128
  const int b_base = blockIdx.x * 16;
  const int o_base = wave * 16;
  const int nlo    = lane & 15;
  const int khalf  = lane >> 4;

  const float* arow = h_in + (size_t)(b_base + nlo) * Hn;
  const float* brow = W_fc + (size_t)(o_base + nlo) * Hn;

  v8f acc = {};
  for (int kc = 0; kc < Hn / 32; ++kc) {
    Frag16 a, b;
    const float* pa = arow + kc * 32 + khalf * 8;
    #pragma unroll
    for (int i = 0; i < 4; ++i) {
      a.u[i]     = pack_bf16x2(pa[2 * i], pa[2 * i + 1]);
      a.u[4 + i] = pack_bf16x2(pa[16 + 2 * i], pa[16 + 2 * i + 1]);
    }
    const float* pb = brow + kc * 32 + khalf * 8;
    #pragma unroll
    for (int i = 0; i < 8; ++i)
      b.u[i] = pack_bf16x2(pb[2 * i + (i >= 4 ? 8 : 0)], pb[2 * i + 1 + (i >= 4 ? 8 : 0)]);
    acc = __builtin_amdgcn_wmma_f32_16x16x32_bf16(
        false, a.v, false, b.v, (short)0, acc, false, false);
  }

  const float bias = b_fc[o_base + nlo];
  #pragma unroll
  for (int r = 0; r < 8; ++r) {
    int m = khalf * 8 + r;
    logits[m * On + o_base + nlo] = acc[r] + bias;
  }
  __syncthreads();

  if (tid < 16) {  // one thread per batch row; trivial work
    const float* row = &logits[tid * On];
    float mx = row[0];
    #pragma unroll 4
    for (int i = 1; i < On; ++i) mx = fmaxf(mx, row[i]);
    float s = 0.f;
    #pragma unroll 4
    for (int i = 0; i < On; ++i) s += __expf(row[i] - mx);
    float inv = 1.f / s;
    float* orow = out + (size_t)(b_base + tid) * On;
    #pragma unroll 4
    for (int i = 0; i < On; ++i) orow[i] = __expf(row[i] - mx) * inv;
  }
}

// ---------------------------------------------------------------------------
extern "C" void kernel_launch(void* const* d_in, const int* in_sizes, int n_in,
                              void* d_out, int out_size, void* d_ws, size_t ws_size,
                              hipStream_t stream) {
  (void)in_sizes; (void)n_in; (void)out_size;
  const float* x    = (const float*)d_in[0];
  // d_in[1] = batchSize scalar (unused; shapes are compile-time)
  const float* W_ih = (const float*)d_in[2];
  const float* b_ih = (const float*)d_in[3];
  const float* u    = (const float*)d_in[4];
  const float* W_fc = (const float*)d_in[5];
  const float* b_fc = (const float*)d_in[6];

  const size_t hbytes = (size_t)Bn * Hn * sizeof(float);       // 512 KB h_last
  const size_t xbytes = (size_t)Bn * Tn * Dn * 2;              // 128 MB bf16 x
  float* hws = (float*)d_ws;
  unsigned short* xb = (unsigned short*)((char*)d_ws + hbytes);

  const bool pre = (ws_size >= hbytes + xbytes);
  if (pre) {
    const int nthreads = Bn * Tn * Dn / 8;  // 8 floats per thread
    convert_bf16_kernel<<<nthreads / 256, 256, 0, stream>>>(x, xb);
    indrnn_scan_kernel<true><<<dim3(Bn / 16, Hn / 16), 32, 0, stream>>>(
        x, xb, W_ih, b_ih, u, hws);
  } else {
    indrnn_scan_kernel<false><<<dim3(Bn / 16, Hn / 16), 32, 0, stream>>>(
        x, xb, W_ih, b_ih, u, hws);
  }
  fc_softmax_kernel<<<Bn / 16, 256, 0, stream>>>(hws, W_fc, b_fc, (float*)d_out);
}
